// GAT_2499670966779
// MI455X (gfx1250) — matrix operand order repaired and checked by reading, not compiled, not run on hardware
//
#include <hip/hip_runtime.h>
#include <hip/hip_bf16.h>

#define N_NODES 6000
#define NPAD    6016
#define D_IN    512
#define HID     256
#define HEADS   8
#define CONCAT  2048
#define NCLS    32
#define NEGINF  (-1e9f)
#define ALPHA   0.2f
#define HROW    40        // padded LDS row stride (halves) for H tiles

typedef __attribute__((ext_vector_type(16))) _Float16 v16h;
typedef __attribute__((ext_vector_type(8)))  float    v8f;
typedef __attribute__((ext_vector_type(4)))  unsigned v4u;
typedef __attribute__((ext_vector_type(8)))  int      v8i;
typedef __attribute__((ext_vector_type(4)))  int      v4i;

union Frag16 { v16h v; uint4 q[2]; _Float16 h[16]; };

// A-fragment (16x32 f16, row-major source): lane l -> row l&15,
// halves k0..k0+7 and k0+16..k0+23 where k0 = kbase + (l>=16 ? 8 : 0).
__device__ __forceinline__ v16h ld_a_frag(const _Float16* A, int lda, int kbase) {
    const int l  = threadIdx.x & 31;
    const int k0 = kbase + ((l >> 4) << 3);
    const _Float16* p = A + (size_t)(l & 15) * lda + k0;
    Frag16 f;
    f.q[0] = *(const uint4*)p;
    f.q[1] = *(const uint4*)(p + 16);
    return f.v;
}

// B-fragment (32x16 f16) from Bt stored [N][K] row-major: lane l -> column
// n = l&15; halves k0..k0+7 and k0+16..k0+23 where k0 = kbase + (l>=16 ? 8 : 0).
__device__ __forceinline__ v16h ld_b_frag(const _Float16* Bt, int ldb, int kbase) {
    const int l = threadIdx.x & 31;
    const _Float16* p = Bt + (size_t)(l & 15) * ldb + kbase + ((l >> 4) << 3);
    Frag16 f;
    f.q[0] = *(const uint4*)p;
    f.q[1] = *(const uint4*)(p + 16);
    return f.v;
}

__device__ __forceinline__ v8f wmma16(v16h a, v16h b, v8f c) {
    return __builtin_amdgcn_wmma_f32_16x16x32_f16(false, a, false, b,
                                                  (short)0, c, false, false);
}

// ---- Tensor Data Mover: 2-D f16 tile (tile_d0 x tile_d1) global -> LDS ----
// Builds the D# per CDNA5 ISA 8.3/8.4. Padded LDS rows: every 16 DWORDs
// (64B = one 32-elem f16 row) insert 4 DWORDs -> 80B row stride in LDS.
__device__ __forceinline__ void tdm_load_2d_f16(unsigned lds_addr, const void* gptr,
                                                unsigned tensor_d0, unsigned tensor_d1,
                                                unsigned stride0,
                                                unsigned tile_d0, unsigned tile_d1) {
    unsigned long long ga = (unsigned long long)(size_t)gptr;
    v4u g0;
    g0[0] = 1u;                                   // count=1, is_restore=0, no gather
    g0[1] = lds_addr;                             // LDS byte address
    g0[2] = (unsigned)ga;                         // global addr [31:0]
    g0[3] = (unsigned)((ga >> 32) & 0x1FFFFFFu) | (2u << 30);  // addr[56:32] | type=2
    v8i g1;
    // data_size=1 (2B) <<16 | pad_enable<<20 | pad_interval=3 (16 DW) <<22
    // | pad_amount=3 (4 DW) <<25
    g1[0] = (int)((1u << 16) | (1u << 20) | (3u << 22) | (3u << 25));
    g1[1] = (int)((tensor_d0 & 0xFFFFu) << 16);                 // tensor_dim0 lo
    g1[2] = (int)((tensor_d0 >> 16) | ((tensor_d1 & 0xFFFFu) << 16));
    g1[3] = (int)((tensor_d1 >> 16) | ((tile_d0 & 0xFFFFu) << 16));
    g1[4] = (int)(tile_d1 & 0xFFFFu);                           // tile_dim1, tile_dim2=0
    g1[5] = (int)stride0;                                       // tensor_dim0_stride lo32
    g1[6] = 0;
    g1[7] = 0;
    v4i z4 = {0, 0, 0, 0};
#if __clang_major__ >= 23
    v8i z8 = {0, 0, 0, 0, 0, 0, 0, 0};
    __builtin_amdgcn_tensor_load_to_lds(g0, g1, z4, z4, z8, 0);
#else
    __builtin_amdgcn_tensor_load_to_lds(g0, g1, z4, z4, 0);
#endif
}

// ---------------- conversion kernels ----------------
__global__ void k_cvt(const float* __restrict__ in, _Float16* __restrict__ out, long n) {
    long i = (long)blockIdx.x * blockDim.x + threadIdx.x;
    if (i < n) out[i] = (_Float16)in[i];
}

__global__ void k_cvt_w0t(const float* __restrict__ W0, _Float16* __restrict__ W0t) {
    long i = (long)blockIdx.x * blockDim.x + threadIdx.x;
    const long total = (long)HEADS * D_IN * HID;
    if (i >= total) return;
    int h   = (int)(i / ((long)HID * D_IN));
    int rem = (int)(i % ((long)HID * D_IN));
    int n   = rem / D_IN;
    int k   = rem % D_IN;
    W0t[i] = (_Float16)W0[((size_t)h * D_IN + k) * HID + n];
}

__global__ void k_cvt_wct(const float* __restrict__ Wc, _Float16* __restrict__ Wct) {
    long i = (long)blockIdx.x * blockDim.x + threadIdx.x;
    if (i >= (long)NCLS * CONCAT) return;
    int n = (int)(i / CONCAT);
    int k = (int)(i % CONCAT);
    Wct[i] = (_Float16)Wc[(size_t)k * NCLS + n];
}

__global__ void k_zero_pads(_Float16* __restrict__ Ht, _Float16* __restrict__ HcT) {
    int t = blockIdx.x * blockDim.x + threadIdx.x;
    if (t < HEADS * HID * 16) {
        Ht[(size_t)(t >> 4) * NPAD + N_NODES + (t & 15)] = (_Float16)0.f;
    } else {
        int t2 = t - HEADS * HID * 16;
        if (t2 < NCLS * 16)
            HcT[(size_t)(t2 >> 4) * NPAD + N_NODES + (t2 & 15)] = (_Float16)0.f;
    }
}

// ---------------- generic batched WMMA GEMM, transposed f16 output ----------------
// C^T[n][m] = sum_k A[m][k] * Bt[n][k]
__global__ void k_gemm(const _Float16* __restrict__ A, int lda, long sA,
                       const _Float16* __restrict__ Bt, int ldb, long sB,
                       _Float16* __restrict__ Ct, int ldc, long sC,
                       int K, int Ntiles) {
    const int wid = threadIdx.x >> 5;
    const int nt  = blockIdx.y * (blockDim.x >> 5) + wid;
    if (nt >= Ntiles) return;
    const int b = blockIdx.z;
    A  += (size_t)b * sA + (size_t)(blockIdx.x * 16) * lda;
    Bt += (size_t)b * sB + (size_t)(nt * 16) * ldb;
    Ct += (size_t)b * sC;

    v8f acc = {0.f, 0.f, 0.f, 0.f, 0.f, 0.f, 0.f, 0.f};
    for (int k = 0; k < K; k += 32) {
        v16h a  = ld_a_frag(A, lda, k);
        v16h bb = ld_b_frag(Bt, ldb, k);
        acc = wmma16(a, bb, acc);
    }
    const int l  = threadIdx.x & 31;
    const int n  = nt * 16 + (l & 15);
    const int m0 = blockIdx.x * 16 + ((l >> 4) << 3);
    _Float16* cp = Ct + (size_t)n * ldc + m0;
#pragma unroll
    for (int vg = 0; vg < 8; ++vg) cp[vg] = (_Float16)acc[vg];
}

// ---------------- f1/f2 projections ----------------
__global__ void k_f1f2(const _Float16* __restrict__ HtAll, long strideH,
                       const float* __restrict__ a1, const float* __restrict__ a2,
                       int strideA, float* __restrict__ f1, float* __restrict__ f2,
                       int Fdim) {
    const int h = blockIdx.y;
    const int i = blockIdx.x * blockDim.x + threadIdx.x;
    if (i >= N_NODES) return;
    const _Float16* H = HtAll + (size_t)h * strideH;
    const float* a1h = a1 + (size_t)h * strideA;
    const float* a2h = a2 + (size_t)h * strideA;
    float s1 = 0.f, s2 = 0.f;
    for (int f = 0; f < Fdim; ++f) {
        float v = (float)H[(size_t)f * NPAD + i];
        s1 += v * a1h[f];
        s2 += v * a2h[f];
    }
    f1[(size_t)h * N_NODES + i] = s1;
    f2[(size_t)h * N_NODES + i] = s2;
}

// ---------------- row softmax statistics: (max, 1/sum) ----------------
__global__ void k_stats(const int* __restrict__ adj, const float* __restrict__ f1b,
                        const float* __restrict__ f2b, float2* __restrict__ ms,
                        int set0) {
    const int set = set0 + blockIdx.y;
    const int row = blockIdx.x * 8 + (threadIdx.x >> 5);
    const int l   = threadIdx.x & 31;
    const float f1r = f1b[(size_t)set * N_NODES + row];
    const float* f2r = f2b + (size_t)set * N_NODES;
    const long arow = (long)row * N_NODES;
    float m = -3.0e38f, s = 0.f;
    for (int j = l; j < N_NODES; j += 32) {
        __builtin_prefetch(&adj[arow + j + 2048], 0, 1);   // global_prefetch_b8
        float x = f1r + f2r[j];
        float e = (x > 0.f ? x : ALPHA * x) + (adj[arow + j] > 0 ? 0.f : NEGINF);
        float nm = fmaxf(m, e);
        s = s * __expf(m - nm) + __expf(e - nm);
        m = nm;
    }
#pragma unroll
    for (int off = 16; off > 0; off >>= 1) {
        float om = __shfl_xor(m, off, 32);
        float os = __shfl_xor(s, off, 32);
        float nm = fmaxf(m, om);
        s = s * __expf(m - nm) + os * __expf(om - nm);
        m = nm;
    }
    if (l == 0) ms[(size_t)set * N_NODES + row] = make_float2(m, 1.0f / s);
}

// ---------------- fused attention aggregation, all 8 heads per block ----------------
// TDM stages each head's 32x256 H tile into LDS (padded rows) while the wave
// computes the softmax-probability A-fragment; WMMA consumes LDS B-fragments.
__global__ __launch_bounds__(256) void k_attn_heads(
    const int* __restrict__ adj, const _Float16* __restrict__ Ht,
    const float* __restrict__ f1, const float* __restrict__ f2,
    const float2* __restrict__ ms, _Float16* __restrict__ O1) {
    __shared__ _Float16 s_h[HEADS][HID][HROW];   // 160 KB, TDM-padded rows
    __shared__ int      s_adj[16][32];
    __shared__ float    s_f2[HEADS][32];
    const int tid  = threadIdx.x;
    const int head = tid >> 5;
    const int l    = tid & 31;
    const int r    = l & 15;
    const int hi   = l >> 4;
    const int row0 = blockIdx.x * 16;

    const float  f1r = f1[(size_t)head * N_NODES + row0 + r];
    const float2 msr = ms[(size_t)head * N_NODES + row0 + r];
    const float  mrow = msr.x, inv_s = msr.y;
    const _Float16* Hh = Ht + (size_t)head * HID * NPAD;
    const unsigned lds_base = (unsigned)(size_t)&s_h[head][0][0];

    v8f acc[16];
    const v8f z = {0.f, 0.f, 0.f, 0.f, 0.f, 0.f, 0.f, 0.f};
#pragma unroll
    for (int i = 0; i < 16; ++i) acc[i] = z;

    for (int j0 = 0; j0 < N_NODES; j0 += 32) {
        // 1) kick off async DMA of this head's H tile (k=j0..j0+31, n=0..255)
        tdm_load_2d_f16(lds_base, Hh + j0, NPAD, HID, NPAD, 32, HID);

        // 2) stage adj / f2 tiles (overlapped with the TDM transfer)
#pragma unroll
        for (int t = tid; t < 512; t += 256) {
            int rr = t >> 5, kk = t & 31, j = j0 + kk;
            s_adj[rr][kk] = (j < N_NODES) ? adj[(size_t)(row0 + rr) * N_NODES + j] : 0;
        }
        {
            int hh = tid >> 5, kk = tid & 31, j = j0 + kk;
            s_f2[hh][kk] = (j < N_NODES) ? f2[(size_t)hh * N_NODES + j] : 0.f;
        }
        __syncthreads();

        // 3) softmax probabilities directly in A-fragment layout
        Frag16 pa;
#pragma unroll
        for (int i = 0; i < 16; ++i) {
            int kk = (hi << 3) + (i & 7) + ((i >> 3) << 4);
            float x = f1r + s_f2[head][kk];
            float e = (x > 0.f ? x : ALPHA * x) + (s_adj[r][kk] > 0 ? 0.f : NEGINF);
            pa.h[i] = (_Float16)(__expf(e - mrow) * inv_s);
        }
        __syncthreads();

        // 4) wait for DMA, then 16 WMMAs fed from LDS
        __builtin_amdgcn_s_wait_tensorcnt(0);
#pragma unroll
        for (int nt = 0; nt < 16; ++nt) {
            const _Float16* p = &s_h[head][nt * 16 + r][hi << 3];
            Frag16 bf;
            bf.q[0] = *(const uint4*)p;
            bf.q[1] = *(const uint4*)(p + 16);
            acc[nt] = wmma16(pa.v, bf.v, acc[nt]);
        }
    }

    const int m0 = row0 + (hi << 3);
#pragma unroll
    for (int nt = 0; nt < 16; ++nt) {
        int col = head * HID + nt * 16 + r;
#pragma unroll
        for (int vg = 0; vg < 8; ++vg) {
            float v = acc[nt][vg];
            float e = v > 0.f ? v : (__expf(v) - 1.f);   // ELU
            O1[(size_t)(m0 + vg) * CONCAT + col] = (_Float16)e;
        }
    }
}

// ---------------- classifier attention (F = 32) ----------------
__global__ __launch_bounds__(64) void k_attn_cls(
    const int* __restrict__ adj, const _Float16* __restrict__ HcT,
    const float* __restrict__ f1c, const float* __restrict__ f2c,
    const float2* __restrict__ msc, float* __restrict__ out) {
    __shared__ int   s_adj[16][32];
    __shared__ float s_f2[32];
    const int tid = threadIdx.x;
    const int nt  = tid >> 5;
    const int l   = tid & 31;
    const int r   = l & 15;
    const int hi  = l >> 4;
    const int row0 = blockIdx.x * 16;

    const float  f1r = f1c[row0 + r];
    const float2 msr = msc[row0 + r];
    const float  mrow = msr.x, inv_s = msr.y;

    v8f acc = {0.f, 0.f, 0.f, 0.f, 0.f, 0.f, 0.f, 0.f};
    for (int j0 = 0; j0 < N_NODES; j0 += 32) {
#pragma unroll
        for (int t = tid; t < 512; t += 64) {
            int rr = t >> 5, kk = t & 31, j = j0 + kk;
            s_adj[rr][kk] = (j < N_NODES) ? adj[(size_t)(row0 + rr) * N_NODES + j] : 0;
        }
        if (tid < 32) {
            int j = j0 + tid;
            s_f2[tid] = (j < N_NODES) ? f2c[j] : 0.f;
        }
        __syncthreads();

        Frag16 pa;
#pragma unroll
        for (int i = 0; i < 16; ++i) {
            int kk = (hi << 3) + (i & 7) + ((i >> 3) << 4);
            float x = f1r + s_f2[kk];
            float e = (x > 0.f ? x : ALPHA * x) + (s_adj[r][kk] > 0 ? 0.f : NEGINF);
            pa.h[i] = (_Float16)(__expf(e - mrow) * inv_s);
        }
        __syncthreads();

        v16h bf = ld_b_frag(HcT + (size_t)(nt * 16) * NPAD, NPAD, j0);
        acc = wmma16(pa.v, bf, acc);
    }

    const int m0 = row0 + (hi << 3);
#pragma unroll
    for (int vg = 0; vg < 8; ++vg)
        out[(size_t)(m0 + vg) * NCLS + nt * 16 + r] = acc[vg];
}

// ---------------- host launcher ----------------
extern "C" void kernel_launch(void* const* d_in, const int* in_sizes, int n_in,
                              void* d_out, int out_size, void* d_ws, size_t ws_size,
                              hipStream_t stream) {
    const float* X   = (const float*)d_in[0];
    const int*   adj = (const int*)d_in[1];
    const float* W0  = (const float*)d_in[2];
    const float* a10 = (const float*)d_in[3];
    const float* a20 = (const float*)d_in[4];
    const float* Wc  = (const float*)d_in[5];
    const float* a1c = (const float*)d_in[6];
    const float* a2c = (const float*)d_in[7];
    float* out = (float*)d_out;

    char* ws = (char*)d_ws;
    size_t o = 0;
    auto take = [&](size_t bytes) -> char* {
        char* p = ws + o;
        o = (o + bytes + 255) & ~(size_t)255;
        return p;
    };
    _Float16* Xh    = (_Float16*)take((size_t)N_NODES * D_IN * 2);
    _Float16* W0t   = (_Float16*)take((size_t)HEADS * HID * D_IN * 2);
    _Float16* Wct   = (_Float16*)take((size_t)NCLS * CONCAT * 2);
    _Float16* Ht    = (_Float16*)take((size_t)HEADS * HID * NPAD * 2);
    _Float16* HcT   = (_Float16*)take((size_t)NCLS * NPAD * 2);
    _Float16* O1    = (_Float16*)take((size_t)N_NODES * CONCAT * 2);
    float*    f1buf = (float*)take((size_t)9 * N_NODES * 4);
    float*    f2buf = (float*)take((size_t)9 * N_NODES * 4);
    float2*   ms    = (float2*)take((size_t)9 * N_NODES * 8);

    // 1) precision conversion / transposition
    k_cvt<<<(N_NODES * D_IN + 255) / 256, 256, 0, stream>>>(X, Xh, (long)N_NODES * D_IN);
    k_cvt_w0t<<<(HEADS * D_IN * HID + 255) / 256, 256, 0, stream>>>(W0, W0t);
    k_cvt_wct<<<(NCLS * CONCAT + 255) / 256, 256, 0, stream>>>(Wc, Wct);
    k_zero_pads<<<(HEADS * HID * 16 + NCLS * 16 + 255) / 256, 256, 0, stream>>>(Ht, HcT);

    // 2) per-head feature GEMM: Ht[h] = (X @ W0[h])^T
    k_gemm<<<dim3(N_NODES / 16, 2, HEADS), 256, 0, stream>>>(
        Xh, D_IN, 0L, W0t, D_IN, (long)HID * D_IN,
        Ht, NPAD, (long)HID * NPAD, D_IN, HID / 16);

    // 3) attention logit projections + softmax stats (heads)
    k_f1f2<<<dim3((N_NODES + 255) / 256, HEADS), 256, 0, stream>>>(
        Ht, (long)HID * NPAD, a10, a20, HID, f1buf, f2buf, HID);
    k_stats<<<dim3(N_NODES / 8, HEADS), 256, 0, stream>>>(adj, f1buf, f2buf, ms, 0);

    // 4) fused softmax + P@H + ELU for all 8 heads (TDM-staged H tiles)
    k_attn_heads<<<N_NODES / 16, 256, 0, stream>>>(adj, Ht, f1buf, f2buf, ms, O1);

    // 5) classifier GEMM: HcT = (O1 @ Wc)^T
    k_gemm<<<dim3(N_NODES / 16, 1, 1), 64, 0, stream>>>(
        O1, CONCAT, 0L, Wct, CONCAT, 0L, HcT, NPAD, 0L, CONCAT, NCLS / 16);

    // 6) classifier logits + stats
    k_f1f2<<<dim3((N_NODES + 255) / 256, 1), 256, 0, stream>>>(
        HcT, 0L, a1c, a2c, 0, f1buf + 8 * N_NODES, f2buf + 8 * N_NODES, NCLS);
    k_stats<<<dim3(N_NODES / 8, 1), 256, 0, stream>>>(adj, f1buf, f2buf, ms, 8);

    // 7) final fused attention -> output
    k_attn_cls<<<N_NODES / 16, 64, 0, stream>>>(
        adj, HcT, f1buf + 8 * N_NODES, f2buf + 8 * N_NODES, ms + 8 * N_NODES, out);
}
